// VAE_86114094285161
// MI455X (gfx1250) — matrix operand order repaired and checked by reading, not compiled, not run on hardware
//
#include <hip/hip_runtime.h>
#include <hip/hip_bf16.h>
#include <cstdint>

// ---------------- problem dims ----------------
#define B_SZ   256
#define T_SZ   256
#define H_SZ   1024
#define G3     3072      // 3*H
#define IN_ENC 133
#define IN_ENC_P 160     // padded to multiple of 32
#define IN_DEC 128
#define ZDIM   251
#define LATD   256
#define NGEN   5

typedef __attribute__((ext_vector_type(16))) __bf16 bf16x16;
typedef __attribute__((ext_vector_type(8)))  float  f32x8;

__device__ __forceinline__ uint16_t f2bf(float f) {
  union { float f; uint32_t u; } x; x.f = f;
  uint32_t u = x.u;
  u += 0x7FFFu + ((u >> 16) & 1u);      // round-to-nearest-even
  return (uint16_t)(u >> 16);
}

struct Frag {
  union { bf16x16 v; uint4 q[2]; };
};

// A fragments for 2 M-tiles (16x32 bf16 each, ISA layout): 2x two 16B loads.
__device__ __forceinline__ void load_a2(Frag* __restrict__ f,
                                        const uint16_t* __restrict__ Arow0,
                                        const uint16_t* __restrict__ Arow1, int kb) {
  const uint16_t* a0 = Arow0 + (kb << 5);
  const uint16_t* a1 = Arow1 + (kb << 5);
  f[0].q[0] = *reinterpret_cast<const uint4*>(a0);        // K = koff .. koff+7
  f[0].q[1] = *reinterpret_cast<const uint4*>(a0 + 16);   // K = koff+16 .. +23
  f[1].q[0] = *reinterpret_cast<const uint4*>(a1);
  f[1].q[1] = *reinterpret_cast<const uint4*>(a1 + 16);
}

// 4 B fragments (pre-packed per-lane layout): two 16B loads each.
__device__ __forceinline__ void load_b4(Frag* __restrict__ f,
                                        const uint16_t* const* __restrict__ Bp, int kb) {
#pragma unroll
  for (int j = 0; j < 4; ++j) {
    const uint16_t* bp = Bp[j] + ((size_t)kb << 9);
    f[j].q[0] = *reinterpret_cast<const uint4*>(bp);
    f[j].q[1] = *reinterpret_cast<const uint4*>(bp + 8);
  }
}

// 8 WMMAs: 2 M-tiles x 4 N-tiles, B reused across M.
__device__ __forceinline__ void mma8(f32x8 (* __restrict__ acc)[4],
                                     const Frag* __restrict__ a,
                                     const Frag* __restrict__ b) {
#pragma unroll
  for (int mi = 0; mi < 2; ++mi)
#pragma unroll
    for (int j = 0; j < 4; ++j)
      acc[mi][j] = __builtin_amdgcn_wmma_f32_16x16x32_bf16(
          false, a[mi].v, false, b[j].v, (short)0, acc[mi][j], false, false);
}

// ---------------------------------------------------------------------------
// One wave computes a 32x64 fp32 tile of C = A(bf16,row-major) * Bpk + bias.
// Double-buffered K loop: loads for block k+1 overlap WMMAs for block k.
// ---------------------------------------------------------------------------
__device__ __forceinline__ void gemm_tile32x64(
    const uint16_t* __restrict__ A, int lda,
    const uint16_t* __restrict__ Bpk,
    const float* __restrict__ bias,
    float* __restrict__ C, int ldc,
    uint16_t* __restrict__ Cbf, int ldcbf,
    int K, int mt, int ng, int lane)
{
  const int m0 = mt << 5;                 // 32 rows per wave
  const int n0 = ng << 6;                 // 64 cols per wave
  const int kbCnt = K >> 5;
  const uint16_t* Arow0 = A + (size_t)(m0 + (lane & 15)) * (size_t)lda + ((lane >> 4) << 3);
  const uint16_t* Arow1 = Arow0 + (size_t)16 * (size_t)lda;
  const int nt0 = n0 >> 4;

  const uint16_t* Bp[4];
#pragma unroll
  for (int j = 0; j < 4; ++j)
    Bp[j] = Bpk + (((size_t)(nt0 + j) * kbCnt) << 9) + (lane << 4);

  f32x8 acc[2][4] = {{f32x8{}, f32x8{}, f32x8{}, f32x8{}},
                     {f32x8{}, f32x8{}, f32x8{}, f32x8{}}};

  Frag aE[2], bE[4], aO[2], bO[4];
  load_a2(aE, Arow0, Arow1, 0);
  load_b4(bE, Bp, 0);

  int kb = 0;
  for (; kb + 2 <= kbCnt; kb += 2) {
    // prefetch odd block while even block computes
    load_a2(aO, Arow0, Arow1, kb + 1);
    load_b4(bO, Bp, kb + 1);
    mma8(acc, aE, bE);
    // prefetch next even block while odd block computes
    if (kb + 2 < kbCnt) {
      load_a2(aE, Arow0, Arow1, kb + 2);
      load_b4(bE, Bp, kb + 2);
    }
    mma8(acc, aO, bO);
  }
  if (kb < kbCnt) {          // odd kbCnt tail (fragments already loaded)
    mma8(acc, aE, bE);
  }

  const int col = lane & 15;
  const int rb  = (lane >> 4) << 3;
#pragma unroll
  for (int mi = 0; mi < 2; ++mi) {
#pragma unroll
    for (int j = 0; j < 4; ++j) {
      const int n = n0 + (j << 4) + col;
      const float bv = bias[n];
#pragma unroll
      for (int r = 0; r < 8; ++r) {
        const int m = m0 + (mi << 4) + rb + r;
        const float v = acc[mi][j][r] + bv;
        C[(size_t)m * ldc + n] = v;
        if (Cbf) Cbf[(size_t)m * ldcbf + n] = f2bf(v);
      }
    }
  }
}

#define MTILES (B_SZ / 32)    // 8 waves in M per problem

// Generic single GEMM (used for decoder output projection)
__global__ void __launch_bounds__(128)
k_gemm(const uint16_t* __restrict__ A, int lda,
       const uint16_t* __restrict__ Bpk, const float* __restrict__ bias,
       float* __restrict__ C, int ldc,
       uint16_t* __restrict__ Cbf, int ldcbf, int N, int K)
{
  const int lane = threadIdx.x & 31;
  const int w = blockIdx.x * 4 + (threadIdx.x >> 5);
  const int nGroups = N >> 6;
  if (w >= MTILES * nGroups) return;
  gemm_tile32x64(A, lda, Bpk, bias, C, ldc, Cbf, ldcbf, K, w / nGroups, w % nGroups, lane);
}

// Fused gi + gh GEMM for one GRU step (both outputs are [256 x 3072])
__global__ void __launch_bounds__(128)
k_gru_gemm2(const uint16_t* __restrict__ A1, int lda1,
            const uint16_t* __restrict__ B1, const float* __restrict__ bias1, int K1,
            const uint16_t* __restrict__ A2, int lda2,
            const uint16_t* __restrict__ B2, const float* __restrict__ bias2, int K2,
            float* __restrict__ C1, float* __restrict__ C2)
{
  const int lane = threadIdx.x & 31;
  int w = blockIdx.x * 4 + (threadIdx.x >> 5);
  const int nGroups = G3 >> 6;                   // 48
  const int half = MTILES * nGroups;             // 384 waves each
  if (w < half) {
    gemm_tile32x64(A1, lda1, B1, bias1, C1, G3, nullptr, 0, K1, w / nGroups, w % nGroups, lane);
  } else {
    w -= half;
    gemm_tile32x64(A2, lda2, B2, bias2, C2, G3, nullptr, 0, K2, w / nGroups, w % nGroups, lane);
  }
}

// GRU gate combine: h' = (1-z)*n + z*h  (PyTorch GRUCell math)
__global__ void __launch_bounds__(256)
k_gru_gates(const float* __restrict__ gi, const float* __restrict__ gh,
            float* __restrict__ hf, uint16_t* __restrict__ hbf)
{
  const int idx = blockIdx.x * 256 + threadIdx.x;
  if (idx >= B_SZ * H_SZ) return;
  const int b = idx >> 10, j = idx & 1023;
  const float* gib = gi + (size_t)b * G3;
  const float* ghb = gh + (size_t)b * G3;
  const float r = 1.0f / (1.0f + __expf(-(gib[j] + ghb[j])));
  const float z = 1.0f / (1.0f + __expf(-(gib[H_SZ + j] + ghb[H_SZ + j])));
  const float n = tanhf(gib[2 * H_SZ + j] + r * ghb[2 * H_SZ + j]);
  const float hn = (1.0f - z) * n + z * hf[idx];
  hf[idx] = hn;
  hbf[idx] = f2bf(hn);
}

// mu / logvar heads + reparameterize + concat genre -> zcat [256 x 256]
__global__ void __launch_bounds__(256)
k_heads(const float* __restrict__ hf,
        const float* __restrict__ Wmu, const float* __restrict__ bmu,
        const float* __restrict__ Wlv, const float* __restrict__ blv,
        const float* __restrict__ eps, const float* __restrict__ genre,
        float* __restrict__ mu_out, float* __restrict__ lv_out,
        float* __restrict__ zcat)
{
  const int idx = blockIdx.x * 256 + threadIdx.x;
  if (idx >= B_SZ * LATD) return;
  const int b = idx >> 8, j = idx & 255;
  if (j < ZDIM) {
    const float* h = hf + (size_t)b * H_SZ;
    const float* wm = Wmu + (size_t)j * H_SZ;
    const float* wl = Wlv + (size_t)j * H_SZ;
    float smu = bmu[j], slv = blv[j];
    for (int k = 0; k < H_SZ; ++k) {
      smu = fmaf(h[k], wm[k], smu);
      slv = fmaf(h[k], wl[k], slv);
    }
    mu_out[b * ZDIM + j] = smu;
    lv_out[b * ZDIM + j] = slv;
    zcat[idx] = smu + __expf(0.5f * slv) * eps[b * ZDIM + j];
  } else {
    zcat[idx] = genre[b * NGEN + (j - ZDIM)];
  }
}

// h_dec0 = zcat @ W_out^T + b_out
__global__ void __launch_bounds__(256)
k_hdec0(const float* __restrict__ zcat, const float* __restrict__ Wout,
        const float* __restrict__ bout, float* __restrict__ hf,
        uint16_t* __restrict__ hbf)
{
  const int idx = blockIdx.x * 256 + threadIdx.x;
  if (idx >= B_SZ * H_SZ) return;
  const int b = idx >> 10, j = idx & 1023;
  const float* zr = zcat + (size_t)b * LATD;
  const float* wr = Wout + (size_t)j * LATD;
  float s = bout[j];
  for (int k = 0; k < LATD; ++k) s = fmaf(zr[k], wr[k], s);
  hf[idx] = s;
  hbf[idx] = f2bf(s);
}

// Pack W [N x Ksrc] fp32 (PyTorch: C[m,n] = sum_k A[m,k]*W[n,k]) into the
// per-lane WMMA B-fragment layout (zero-padded to Kp), bf16.
__global__ void __launch_bounds__(256)
k_pack_w(const float* __restrict__ W, uint16_t* __restrict__ out,
         int Ksrc, int Kp, int total)
{
  const int idx = blockIdx.x * 256 + threadIdx.x;
  if (idx >= total) return;
  const int block  = idx >> 9;          // (ntile, kblock)
  const int within = idx & 511;
  const int lane = within >> 4;
  const int i    = within & 15;
  const int kbCnt = Kp >> 5;
  const int nt = block / kbCnt;
  const int kb = block % kbCnt;
  const int n = (nt << 4) + (lane & 15);
  const int k = (kb << 5) + ((lane >> 4) << 3) + (i & 7) + ((i >> 3) << 4);
  const float v = (k < Ksrc) ? W[(size_t)n * Ksrc + k] : 0.0f;
  out[idx] = f2bf(v);
}

// x [B,T,133] fp32 -> xbf [B,T,160] bf16 zero-padded
__global__ void __launch_bounds__(256)
k_convert_x(const float* __restrict__ x, uint16_t* __restrict__ xbf)
{
  const int idx = blockIdx.x * 256 + threadIdx.x;
  if (idx >= B_SZ * T_SZ * IN_ENC_P) return;
  const int bt = idx / IN_ENC_P;
  const int k  = idx - bt * IN_ENC_P;
  const float v = (k < IN_ENC) ? x[(size_t)bt * IN_ENC + k] : 0.0f;
  xbf[idx] = f2bf(v);
}

// ---------------------------------------------------------------------------
extern "C" void kernel_launch(void* const* d_in, const int* in_sizes, int n_in,
                              void* d_out, int out_size, void* d_ws, size_t ws_size,
                              hipStream_t stream)
{
  const float* x     = (const float*)d_in[0];
  const float* genre = (const float*)d_in[1];
  const float* eps   = (const float*)d_in[2];
  const float* eWih  = (const float*)d_in[3];
  const float* eWhh  = (const float*)d_in[4];
  const float* eBih  = (const float*)d_in[5];
  const float* eBhh  = (const float*)d_in[6];
  const float* dWih  = (const float*)d_in[7];
  const float* dWhh  = (const float*)d_in[8];
  const float* dBih  = (const float*)d_in[9];
  const float* dBhh  = (const float*)d_in[10];
  const float* Wmu   = (const float*)d_in[11];
  const float* bmu   = (const float*)d_in[12];
  const float* Wlv   = (const float*)d_in[13];
  const float* blv   = (const float*)d_in[14];
  const float* Wout  = (const float*)d_in[15];
  const float* bout  = (const float*)d_in[16];
  const float* Wdec  = (const float*)d_in[17];
  const float* bdec  = (const float*)d_in[18];

  uint8_t* wsp = (uint8_t*)d_ws;
  auto alloc = [&](size_t bytes) {
    uint8_t* p = wsp;
    wsp += (bytes + 255) & ~(size_t)255;
    return p;
  };
  uint16_t* xbf    = (uint16_t*)alloc((size_t)B_SZ * T_SZ * IN_ENC_P * 2);
  uint16_t* WihE   = (uint16_t*)alloc((size_t)G3 * IN_ENC_P * 2);
  uint16_t* WhhE   = (uint16_t*)alloc((size_t)G3 * H_SZ * 2);
  uint16_t* WihD   = (uint16_t*)alloc((size_t)G3 * IN_DEC * 2);
  uint16_t* WhhD   = (uint16_t*)alloc((size_t)G3 * H_SZ * 2);
  uint16_t* WdecP  = (uint16_t*)alloc((size_t)IN_DEC * H_SZ * 2);
  float*    hf     = (float*)   alloc((size_t)B_SZ * H_SZ * 4);
  uint16_t* hbf    = (uint16_t*)alloc((size_t)B_SZ * H_SZ * 2);
  float*    gi     = (float*)   alloc((size_t)B_SZ * G3 * 4);
  float*    gh     = (float*)   alloc((size_t)B_SZ * G3 * 4);
  uint16_t* notebf = (uint16_t*)alloc((size_t)B_SZ * IN_DEC * 2);
  float*    zcat   = (float*)   alloc((size_t)B_SZ * LATD * 4);

  float* out    = (float*)d_out;                                  // [B,T,128]
  float* mu_out = out + (size_t)B_SZ * T_SZ * IN_DEC;             // [B,251]
  float* lv_out = mu_out + (size_t)B_SZ * ZDIM;                   // [B,251]

  // ---- prep: bf16 convert + weight packing ----
  k_convert_x<<<(B_SZ * T_SZ * IN_ENC_P) / 256, 256, 0, stream>>>(x, xbf);
  k_pack_w<<<(G3 * IN_ENC_P) / 256, 256, 0, stream>>>(eWih, WihE, IN_ENC, IN_ENC_P, G3 * IN_ENC_P);
  k_pack_w<<<(G3 * H_SZ) / 256, 256, 0, stream>>>(eWhh, WhhE, H_SZ, H_SZ, G3 * H_SZ);
  k_pack_w<<<(G3 * IN_DEC) / 256, 256, 0, stream>>>(dWih, WihD, IN_DEC, IN_DEC, G3 * IN_DEC);
  k_pack_w<<<(G3 * H_SZ) / 256, 256, 0, stream>>>(dWhh, WhhD, H_SZ, H_SZ, G3 * H_SZ);
  k_pack_w<<<(IN_DEC * H_SZ) / 256, 256, 0, stream>>>(Wdec, WdecP, H_SZ, H_SZ, IN_DEC * H_SZ);

  hipMemsetAsync(hf, 0, (size_t)B_SZ * H_SZ * 4, stream);
  hipMemsetAsync(hbf, 0, (size_t)B_SZ * H_SZ * 2, stream);

  const int gemm2Blocks = (2 * MTILES * (G3 >> 6)) / 4;   // 768 waves / 4 per block
  const int gateBlocks  = (B_SZ * H_SZ) / 256;

  // ---- encoder recurrence ----
  for (int t = 0; t < T_SZ; ++t) {
    k_gru_gemm2<<<gemm2Blocks, 128, 0, stream>>>(
        xbf + (size_t)t * IN_ENC_P, T_SZ * IN_ENC_P, WihE, eBih, IN_ENC_P,
        hbf, H_SZ, WhhE, eBhh, H_SZ, gi, gh);
    k_gru_gates<<<gateBlocks, 256, 0, stream>>>(gi, gh, hf, hbf);
  }

  // ---- latent heads + decoder init ----
  k_heads<<<(B_SZ * LATD) / 256, 256, 0, stream>>>(hf, Wmu, bmu, Wlv, blv,
                                                   eps, genre, mu_out, lv_out, zcat);
  k_hdec0<<<(B_SZ * H_SZ) / 256, 256, 0, stream>>>(zcat, Wout, bout, hf, hbf);
  hipMemsetAsync(notebf, 0, (size_t)B_SZ * IN_DEC * 2, stream);

  // ---- decoder recurrence ----
  const int noteBlocks = (MTILES * (IN_DEC >> 6) + 3) / 4;  // 16 waves -> 4 blocks
  for (int t = 0; t < T_SZ; ++t) {
    k_gru_gemm2<<<gemm2Blocks, 128, 0, stream>>>(
        notebf, IN_DEC, WihD, dBih, IN_DEC,
        hbf, H_SZ, WhhD, dBhh, H_SZ, gi, gh);
    k_gru_gates<<<gateBlocks, 256, 0, stream>>>(gi, gh, hf, hbf);
    k_gemm<<<noteBlocks, 128, 0, stream>>>(
        hbf, H_SZ, WdecP, bdec,
        out + (size_t)t * IN_DEC, T_SZ * IN_DEC,
        notebf, IN_DEC, IN_DEC, H_SZ);
  }
}
